// LSTMModel_73306501808958
// MI455X (gfx1250) — compile-verified, matrix operand order checked by reading, not run on hardware
//
#include <hip/hip_runtime.h>
#include <cstdint>
#include <cstddef>

// ---------------------------------------------------------------------------
// Persistent fused 2-layer LSTM for MI455X (gfx1250, wave32, WMMA).
//
//   T=4096, B=512, F=10, H=20, gates=4H=80.
//   grid = 32 blocks x 32 threads (1 wave per block; WMMA is wave-scoped).
//   Each wave owns a 16-row batch tile and runs the whole time loop, fusing
//   layer0 -> layer1 -> FC; h/c state never touches global memory.
//
//   Combined-K trick: one A fragment carries  x(t) at K=0..9,
//   h0 at K=10..29, and a constant 1.0 at K=30.  The matching B fragment
//   packs [W_ih0 | W_hh0 | bias0], so layer-0 gates need a SINGLE
//   v_wmma_f32_16x16x32_f16 per 16-column tile with C = 0.
//   Layer 1 reuses the same LDS read (x columns are 0 in LDS) against
//   [0 | W_ih1 | bias1], plus one chained WMMA with W_hh1 against h1.
//
//   Fragment maps (assumed per ISA 7.12.2):
//     A: row = lane&15; element e -> K = (e<8 ? kb+e : 16+kb+e-8),
//        kb = (lane<16 ? 0 : 8).
//     B: column = lane&15; same K map.
//     C/D: column = lane&15; rows (lane<16 ? 0..7 : 8..15).
// ---------------------------------------------------------------------------

typedef __attribute__((ext_vector_type(16))) _Float16 v16h;
typedef __attribute__((ext_vector_type(8)))  _Float16 v8h;
typedef __attribute__((ext_vector_type(8)))  float    v8f;

#define TT 4096
#define BB 512
#define FF 10
#define HH 20
#define G4 80

__device__ __forceinline__ float sigm(float x) {
    return 1.0f / (1.0f + __expf(-x));
}
__device__ __forceinline__ float tanh_fast(float x) {
    float e = __expf(2.0f * x);
    return (e - 1.0f) / (e + 1.0f);
}
__device__ __forceinline__ v8f zero8() {
    v8f v;
#pragma unroll
    for (int i = 0; i < 8; ++i) v[i] = 0.0f;
    return v;
}

__global__ __launch_bounds__(32)
void lstm2_fused_wmma(const float* __restrict__ x,
                      const float* __restrict__ w_ih0, const float* __restrict__ w_hh0,
                      const float* __restrict__ b_ih0, const float* __restrict__ b_hh0,
                      const float* __restrict__ w_ih1, const float* __restrict__ w_hh1,
                      const float* __restrict__ b_ih1, const float* __restrict__ b_hh1,
                      const float* __restrict__ w_fc,  const float* __restrict__ b_fc,
                      float* __restrict__ out)
{
    // Per-wave LDS staging (one wave per workgroup).
    __shared__ __align__(16) float    gatesT[G4][16];   // [gate col][batch row]
    __shared__ __align__(16) _Float16 comb[16][32];     // cols 0..9 unused(0), 10..29 h0, 30 = 1.0
    __shared__ __align__(16) _Float16 h1A[16][32];      // cols 0..19 h1, rest 0

    const int  lane   = threadIdx.x;          // 0..31
    const int  l16    = lane & 15;
    const bool hiHalf = (lane >= 16);
    const int  kb     = hiHalf ? 8 : 0;       // fragment K base for this lane
    const int  rbase  = hiHalf ? 8 : 0;       // C/D row base for this lane
    const int  b0     = blockIdx.x * 16;      // batch tile origin

    // ---- init LDS: zero both h buffers, then plant the bias-1 slot ----
    {
        uint32_t* p0 = (uint32_t*)&comb[0][0];
        uint32_t* p1 = (uint32_t*)&h1A[0][0];
#pragma unroll
        for (int i = lane; i < 16 * 32 / 2; i += 32) { p0[i] = 0u; p1[i] = 0u; }
        if (lane < 16) comb[lane][30] = (_Float16)1.0f;   // K=30 constant-one
    }

    // ---- pre-pack weights(+bias) into WMMA B-fragment registers (once) ----
    v16h comb0B[5], wih1B[5], whh1B[5];
#pragma unroll
    for (int j = 0; j < 5; ++j) {
        const int col = 16 * j + l16;                    // gate column 0..79
        const float bs0 = b_ih0[col] + b_hh0[col];
        const float bs1 = b_ih1[col] + b_hh1[col];
#pragma unroll
        for (int e = 0; e < 16; ++e) {
            const int K = (e < 8) ? (kb + e) : (16 + kb + (e - 8));
            float w0, w1;
            if (K < FF)            w0 = w_ih0[col * FF + K];
            else if (K < FF + HH)  w0 = w_hh0[col * HH + (K - FF)];
            else if (K == 30)      w0 = bs0;
            else                   w0 = 0.0f;
            if (K >= FF && K < FF + HH) w1 = w_ih1[col * HH + (K - FF)];
            else if (K == 30)           w1 = bs1;
            else                        w1 = 0.0f;
            comb0B[j][e] = (_Float16)w0;
            wih1B[j][e]  = (_Float16)w1;
            whh1B[j][e]  = (_Float16)((K < HH) ? w_hh1[col * HH + K] : 0.0f);
        }
    }

    // elementwise ownership: lane -> batch row em, hidden cols en0..en0+9
    const int em  = lane >> 1;
    const int en0 = (lane & 1) * 10;

    float c0r[10], c1r[10];
#pragma unroll
    for (int p = 0; p < 10; ++p) { c0r[p] = 0.0f; c1r[p] = 0.0f; }

    v16h h1frag;
#pragma unroll
    for (int e = 0; e < 16; ++e) h1frag[e] = (_Float16)0.0f;

    const v8f zc = zero8();      // C = 0 for the first WMMA of each group

    // x addressing: x[b][t][f], batch-major. Lane row = l16.
    // lanes 0..15 fetch f0..7 of their row, lanes 16..31 fetch f8..9.
    const size_t xrow = (size_t)(b0 + l16) * (size_t)TT;

    // preload t = 0
    float xv[8];
    {
        const float* px = x + xrow * FF;
        if (!hiHalf) {
#pragma unroll
            for (int q = 0; q < 4; ++q) {
                float2 v = *(const float2*)(px + 2 * q);
                xv[2 * q] = v.x; xv[2 * q + 1] = v.y;
            }
        } else {
            float2 v = *(const float2*)(px + 8);
            xv[0] = v.x; xv[1] = v.y;
#pragma unroll
            for (int q = 2; q < 8; ++q) xv[q] = 0.0f;
        }
    }

    for (int t = 0; t < TT; ++t) {
        // ---- combined A fragment for layer0: x(t) from regs, h0(t-1)+1.0 from LDS ----
        v8h xh;
#pragma unroll
        for (int e = 0; e < 8; ++e) xh[e] = (_Float16)xv[e];
        v16h fragA;
        {
            v8h cA = *(const v8h*)&comb[l16][kb];        // lanes>=16: cols 8..15 (f8,f9 slots + h0..5)
            v8h cB = *(const v8h*)&comb[l16][16 + kb];   // h6..13 / h14..19 + bias-1
#pragma unroll
            for (int e = 0; e < 8; ++e) {
                // lanes<16: K0..7 = x f0..7 (regs). lanes>=16: K8,9 = x f8,f9 (regs), K10..15 = h (LDS)
                fragA[e]     = hiHalf ? ((e < 2) ? xh[e] : cA[e]) : xh[e];
                fragA[8 + e] = cB[e];
            }
        }

        // ---- prefetch + issue loads for the next timestep ----
        const int tn = (t + 1 < TT) ? (t + 1) : (TT - 1);
        if (t + 8 < TT) __builtin_prefetch(x + (xrow + (size_t)(t + 8)) * FF, 0, 1);
        float xnv[8];
        {
            const float* pnx = x + (xrow + (size_t)tn) * FF;
            if (!hiHalf) {
#pragma unroll
                for (int q = 0; q < 4; ++q) {
                    float2 v = *(const float2*)(pnx + 2 * q);
                    xnv[2 * q] = v.x; xnv[2 * q + 1] = v.y;
                }
            } else {
                float2 v = *(const float2*)(pnx + 8);
                xnv[0] = v.x; xnv[1] = v.y;
#pragma unroll
                for (int q = 2; q < 8; ++q) xnv[q] = 0.0f;
            }
        }

        // ============ LAYER 0: gates = [x|h0|1] * [Wih0|Whh0|b]^T  (1 WMMA/tile) ============
        {
#pragma unroll
            for (int j = 0; j < 5; ++j) {
                v8f acc = __builtin_amdgcn_wmma_f32_16x16x32_f16(
                    false, fragA, false, comb0B[j], (short)0, zc, false, false);
                const int col = 16 * j + l16;
                *(float4*)&gatesT[col][rbase]     = make_float4(acc[0], acc[1], acc[2], acc[3]);
                *(float4*)&gatesT[col][rbase + 4] = make_float4(acc[4], acc[5], acc[6], acc[7]);
            }
        }
        // LDS ops from one wave complete in order -> safe to read back.
#pragma unroll
        for (int p = 0; p < 10; ++p) {
            const int n = en0 + p;
            const float ig = sigm(gatesT[n     ][em]);
            const float fg = sigm(gatesT[n + 20][em]);
            const float gg = tanh_fast(gatesT[n + 40][em]);
            const float og = sigm(gatesT[n + 60][em]);
            const float c  = fg * c0r[p] + ig * gg;
            c0r[p] = c;
            comb[em][FF + n] = (_Float16)(og * tanh_fast(c));   // h0(t) -> K=10..29
        }

        // ---- layer1 input fragment: pure LDS read (x cols are 0, bias-1 at K30) ----
        v16h fragB;
        {
            v8h cA = *(const v8h*)&comb[l16][kb];
            v8h cB = *(const v8h*)&comb[l16][16 + kb];
#pragma unroll
            for (int e = 0; e < 8; ++e) { fragB[e] = cA[e]; fragB[8 + e] = cB[e]; }
        }

        // ============ LAYER 1: gates = [0|h0|1]*[0|Wih1|b]^T + h1*Whh1^T ============
        {
#pragma unroll
            for (int j = 0; j < 5; ++j) {
                v8f acc = __builtin_amdgcn_wmma_f32_16x16x32_f16(
                    false, fragB, false, wih1B[j], (short)0, zc, false, false);
                acc = __builtin_amdgcn_wmma_f32_16x16x32_f16(
                    false, h1frag, false, whh1B[j], (short)0, acc, false, false);
                const int col = 16 * j + l16;
                *(float4*)&gatesT[col][rbase]     = make_float4(acc[0], acc[1], acc[2], acc[3]);
                *(float4*)&gatesT[col][rbase + 4] = make_float4(acc[4], acc[5], acc[6], acc[7]);
            }
        }
#pragma unroll
        for (int p = 0; p < 10; ++p) {
            const int n = en0 + p;
            const float ig = sigm(gatesT[n     ][em]);
            const float fg = sigm(gatesT[n + 20][em]);
            const float gg = tanh_fast(gatesT[n + 40][em]);
            const float og = sigm(gatesT[n + 60][em]);
            const float c  = fg * c1r[p] + ig * gg;
            c1r[p] = c;
            h1A[em][n] = (_Float16)(og * tanh_fast(c));          // h1(t) -> K=0..19
        }
        {
            v8h cA = *(const v8h*)&h1A[l16][kb];
            v8h cB = *(const v8h*)&h1A[l16][16 + kb];
#pragma unroll
            for (int e = 0; e < 8; ++e) { h1frag[e] = cA[e]; h1frag[8 + e] = cB[e]; }
        }

#pragma unroll
        for (int q = 0; q < 8; ++q) xv[q] = xnv[q];
    }

    // ---- FC epilogue: out[b] = h1_last[b,:] . w_fc + b_fc ----
    // lane -> row em, cols en0..en0+9 ; pair (2m, 2m+1) covers the 20 cols.
    float fc = 0.0f;
#pragma unroll
    for (int p = 0; p < 10; ++p)
        fc += (float)h1A[em][en0 + p] * w_fc[en0 + p];
    const float tot = fc + __shfl_xor(fc, 1, 32);
    if ((lane & 1) == 0) out[b0 + em] = tot + b_fc[0];
}

extern "C" void kernel_launch(void* const* d_in, const int* in_sizes, int n_in,
                              void* d_out, int out_size, void* d_ws, size_t ws_size,
                              hipStream_t stream) {
    (void)in_sizes; (void)n_in; (void)out_size; (void)d_ws; (void)ws_size;
    const float* x     = (const float*)d_in[0];
    const float* w_ih0 = (const float*)d_in[1];
    const float* w_hh0 = (const float*)d_in[2];
    const float* b_ih0 = (const float*)d_in[3];
    const float* b_hh0 = (const float*)d_in[4];
    const float* w_ih1 = (const float*)d_in[5];
    const float* w_hh1 = (const float*)d_in[6];
    const float* b_ih1 = (const float*)d_in[7];
    const float* b_hh1 = (const float*)d_in[8];
    const float* w_fc  = (const float*)d_in[9];
    const float* b_fc  = (const float*)d_in[10];
    float* out = (float*)d_out;

    // 32 batch tiles of 16 rows; one wave (32 threads) per workgroup so each
    // wave gets a WGP to itself (single-wave WG => barriers are NOPs).
    lstm2_fused_wmma<<<dim3(BB / 16), dim3(32), 0, stream>>>(
        x, w_ih0, w_hh0, b_ih0, b_hh0,
        w_ih1, w_hh1, b_ih1, b_hh1,
        w_fc, b_fc, out);
}
